// QuantumNATHybrid_65481071395552
// MI455X (gfx1250) — matrix operand order repaired and checked by reading, not compiled, not run on hardware
//
#include <hip/hip_runtime.h>
#include <hip/hip_bf16.h>
#include <stdint.h>

typedef __attribute__((ext_vector_type(16))) _Float16 v16h;
typedef __attribute__((ext_vector_type(8)))  _Float16 v8h;
typedef __attribute__((ext_vector_type(8)))  float    v8f;
typedef __attribute__((ext_vector_type(4)))  unsigned int v4u;
typedef __attribute__((ext_vector_type(4)))  int      v4i;
typedef __attribute__((ext_vector_type(8)))  int      v8i;

#define BATCH 8192
#define N_OPS 50

#if defined(__AMDGCN__) && __has_builtin(__builtin_amdgcn_tensor_load_to_lds)
#define HAVE_TDM 1
#else
#define HAVE_TDM 0
#endif

struct GateList { int t[N_OPS]; int w0[N_OPS]; int w1[N_OPS]; };

#if HAVE_TDM
// ---------------------------------------------------------------------------
// TDM helper: 2-D tensor_load_to_lds. data_size = 2 bytes (f16).
//   tensor_dim0 = valid elements along dim0 (reads beyond -> zero),
//   tile = tile0 x tile1, row stride in elements. 2-D only: groups 2/3 zero.
// D# bitfields per CDNA5 ISA.
// ---------------------------------------------------------------------------
__device__ __forceinline__ void tdm_load_2d(uint32_t lds_off, const void* gaddr,
                                            uint32_t dim0_valid, uint32_t dim1,
                                            uint32_t tile0, uint32_t tile1,
                                            uint32_t row_stride_elems) {
  uint64_t ga = (uint64_t)(uintptr_t)gaddr;
  v4u g0;
  g0.x = 1u;                                    // count=1, user descriptor
  g0.y = lds_off;                               // LDS byte address
  g0.z = (uint32_t)ga;                          // global_addr[31:0]
  g0.w = (uint32_t)(ga >> 32) | (2u << 30);     // global_addr[56:32] | type=2
  v8i g1;
  g1[0] = (int)(1u << 16);                      // workgroup_mask=0, data_size=1 (2B)
  g1[1] = (int)((dim0_valid & 0xFFFFu) << 16);  // tensor_dim0[15:0] @ bits 63:48
  g1[2] = (int)((dim0_valid >> 16) | (dim1 << 16)); // dim0[31:16] | tensor_dim1[15:0]
  g1[3] = (int)(tile0 << 16);                   // tensor_dim1[31:16]=0 | tile_dim0
  g1[4] = (int)tile1;                           // tile_dim1 | tile_dim2=0
  g1[5] = (int)row_stride_elems;                // tensor_dim0_stride[31:0]
  g1[6] = 0;                                    // stride[47:32] | dim1_stride[15:0]
  g1[7] = 0;
  v4i z4 = {0, 0, 0, 0};
#if __clang_major__ >= 23
  v8i z8 = {0, 0, 0, 0, 0, 0, 0, 0};
  __builtin_amdgcn_tensor_load_to_lds(g0, g1, z4, z4, z8, 0);
#else
  __builtin_amdgcn_tensor_load_to_lds(g0, g1, z4, z4, 0);
#endif
}
#endif  // HAVE_TDM

// ---------------------------------------------------------------------------
// Kernel 0: pre-convert fc1_w (f32 [64][784]) into f16 B-fragment order:
// wfrag[kc][sub][lane][h], kc<25 (K padded 784->800), sub = n/16,
// lane = (n&15) + 16*(kk>>4), h = kk&15.  Each K-chunk is a contiguous 4 KB
// blob -> TDM-friendly.
// ---------------------------------------------------------------------------
__global__ void __launch_bounds__(256) k0_wswz(const float* __restrict__ w1,
                                               _Float16* __restrict__ wfrag) {
  int e = blockIdx.x * 256 + threadIdx.x;
  if (e >= 25 * 4 * 32 * 16) return;
  int h    = e & 15;
  int lane = (e >> 4) & 31;
  int sub  = (e >> 9) & 3;
  int kc   = e >> 11;
  int k = kc * 32 + ((lane >> 4) * 16 + h);
  int n = sub * 16 + (lane & 15);
  float v = (k < 784) ? w1[n * 784 + k] : 0.0f;
  wfrag[e] = (_Float16)v;
}

// ---------------------------------------------------------------------------
// Kernel 1: conv1 (1->8, 3x3 SAME) + ReLU + 2x2 maxpool.  Output f16 NCHW.
// Pure VALU: K=9 with 1 input channel is too thin for WMMA; stage is bound by
// streaming x (25.7 MB).
// ---------------------------------------------------------------------------
__global__ void __launch_bounds__(256) k1_conv1(const float* __restrict__ x,
                                                const float* __restrict__ w,
                                                const float* __restrict__ bias,
                                                _Float16* __restrict__ h1) {
  int id = blockIdx.x * 256 + threadIdx.x;            // ((b*8+c)*14+oy)*14+ox
  if (id >= BATCH * 8 * 14 * 14) return;
  int ox = id % 14, t = id / 14;
  int oy = t % 14;  t /= 14;
  int c  = t % 8;
  int b  = t / 8;
  float wc[9];
#pragma unroll
  for (int i = 0; i < 9; ++i) wc[i] = w[c * 9 + i];
  const float bb = bias[c];
  const float* xb = x + b * 784;
  __builtin_prefetch(xb + 784, 0, 0);                 // next image (speculative)
  float mx = 0.0f;                                    // relu floor
#pragma unroll
  for (int py = 0; py < 2; ++py)
#pragma unroll
    for (int px = 0; px < 2; ++px) {
      int Y = 2 * oy + py, X = 2 * ox + px;
      float s = bb;
#pragma unroll
      for (int dy = 0; dy < 3; ++dy) {
        int iy = Y + dy - 1;
        if (iy < 0 || iy > 27) continue;
#pragma unroll
        for (int dx = 0; dx < 3; ++dx) {
          int ix = X + dx - 1;
          if (ix < 0 || ix > 27) continue;
          s += wc[dy * 3 + dx] * xb[iy * 28 + ix];
        }
      }
      mx = s > mx ? s : mx;
    }
  h1[id] = (_Float16)mx;
}

// ---------------------------------------------------------------------------
// Kernel 2: conv2 as implicit GEMM on WMMA.
//   M = B*196 im2col rows, N = 16 out channels, K = 72 (8 ch * 9 taps) -> 96.
// One wave per 16x16 tile, 3 x v_wmma_f32_16x16x32_f16, fused bias+ReLU,
// output f16 NCHW (pre-pool).  The im2col gather (3-element runs + border
// zeros) doesn't map to dense TDM tiles, so A is staged pre-swizzled to the
// ISA fragment layout -> each lane fragment load is one aligned 32B ds read.
// ---------------------------------------------------------------------------
__global__ void __launch_bounds__(128) k2_conv2_wmma(const _Float16* __restrict__ h1,
                                                     const float* __restrict__ w2,
                                                     const float* __restrict__ b2,
                                                     _Float16* __restrict__ h2) {
  __shared__ __align__(32) _Float16 Wf[3][32][16];     // [kchunk][lane][half]
  __shared__ __align__(32) _Float16 Af[4][3][32][16];  // per-wave A fragments
  const int tid  = threadIdx.x;
  const int lane = tid & 31;
  const int wv   = tid >> 5;

  // Stage weights in B-fragment order: lane = n + 16*((kk>>3)&1), h = (kk&7) + (kk>=16 ? 8 : 0)
  for (int e = tid; e < 96 * 16; e += 128) {
    int k = e >> 4, n = e & 15;
    float v = (k < 72) ? w2[n * 72 + k] : 0.0f;        // conv2_w[n][i][dy][dx]
    int kk = k & 31, ch = k >> 5;
    Wf[ch][n + (((kk >> 3) & 1) << 4)][(kk & 7) + ((kk >= 16) ? 8 : 0)] = (_Float16)v;
  }

  // Stage im2col A tile in A-fragment order:
  //   lane = m + 16*((kk>>3)&1),  h = (kk&7) + (kk>=16 ? 8 : 0)
  const int tile = blockIdx.x * 4 + wv;
  const int r0   = tile * 16;
  for (int e = lane; e < 16 * 96; e += 32) {
    int m = e / 96, k = e - m * 96;
    float v = 0.0f;
    if (k < 72) {
      int r = r0 + m;
      int b = r / 196, p = r - b * 196;
      int y = p / 14,  x = p - y * 14;
      int i = k / 9,   tp = k - i * 9;
      int dy = tp / 3, dx = tp - dy * 3;
      int iy = y + dy - 1, ix = x + dx - 1;
      if (iy >= 0 && iy < 14 && ix >= 0 && ix < 14)
        v = (float)h1[((b * 8 + i) * 14 + iy) * 14 + ix];
    }
    int kk = k & 31, ch = k >> 5;
    Af[wv][ch][m + (((kk >> 3) & 1) << 4)][(kk & 7) + ((kk >= 16) ? 8 : 0)] = (_Float16)v;
  }
  __syncthreads();

  v8f acc = {};
#pragma unroll
  for (int ch = 0; ch < 3; ++ch) {
    v16h a  = *(const v16h*)(&Af[wv][ch][lane][0]);
    v16h bm = *(const v16h*)(&Wf[ch][lane][0]);
    acc = __builtin_amdgcn_wmma_f32_16x16x32_f16(false, a, false, bm,
                                                 (short)0, acc, false, false);
  }

  // D layout: n = lane&15, m = 8*(lane>>4) + v
  const int n    = lane & 15;
  const float bb = b2[n];
  const int mb   = (lane >> 4) * 8;
#pragma unroll
  for (int v = 0; v < 8; ++v) {
    int r = r0 + mb + v;
    int b = r / 196, p = r - b * 196;
    int y = p / 14,  x = p - y * 14;
    float o = acc[v] + bb;
    o = o > 0.0f ? o : 0.0f;
    h2[((b * 16 + n) * 14 + y) * 14 + x] = (_Float16)o;
  }
}

// ---------------------------------------------------------------------------
// Kernel 3: 2x2 maxpool + flatten into FC1's A matrix [8192][784] f16.
// Flat index = b*784 + c*49 + y*7 + x  (matches NCHW reshape).
// ---------------------------------------------------------------------------
__global__ void __launch_bounds__(256) k3_pool(const _Float16* __restrict__ h2,
                                               _Float16* __restrict__ a3) {
  int id = blockIdx.x * 256 + threadIdx.x;
  if (id >= BATCH * 784) return;
  int x = id % 7, t = id / 7;
  int y = t % 7;  t /= 7;
  int c = t % 16;
  int b = t / 16;
  const _Float16* base = h2 + ((b * 16 + c) * 14 + 2 * y) * 14 + 2 * x;
  float v0 = (float)base[0],  v1 = (float)base[1];
  float v2 = (float)base[14], v3 = (float)base[15];
  a3[id] = (_Float16)fmaxf(fmaxf(v0, v1), fmaxf(v2, v3));
}

// ---------------------------------------------------------------------------
// Kernel 4: FC1 [8192,784] x [784,64] on WMMA + bias + ReLU, f32 out.
// Block = 4 waves sharing one 16-row M tile; wave wv owns N cols [16wv,16wv+16).
// 25 K-chunks of 32 (784 zero-padded to 800).
// TDM path: wave 0 DMAs the A tile (2-D descriptor: 16 rows x 64B, stride
// 1568B, last chunk clipped via tensor_dim0 -> OOB zeros) and the pre-swizzled
// 4KB W chunk into ping-pong LDS buffers, pipelined two-deep on TENSORcnt.
// A-fragments read the *linear* TDM tile: per lane the 16x32 f16 A layout is
// two contiguous 16B runs -> two ds_load_b128 + shuffle.
// ---------------------------------------------------------------------------
__global__ void __launch_bounds__(128) k4_fc1_wmma(const _Float16* __restrict__ a3,
                                                   const float* __restrict__ w1,
                                                   const _Float16* __restrict__ wfrag,
                                                   const float* __restrict__ b1,
                                                   float* __restrict__ out) {
  const int tid = threadIdx.x, lane = tid & 31, wv = tid >> 5;
  const int mtile = blockIdx.x;                        // 512 tiles
  v8f acc = {};
#if HAVE_TDM
  __shared__ __align__(32) _Float16 At[2][16][32];     // linear A tiles (ping-pong)
  __shared__ __align__(32) _Float16 Wt[2][4][32][16];  // B fragments (ping-pong)
  (void)w1;
  const _Float16* abase = a3 + (size_t)mtile * 16 * 784;
  if (wv == 0) {                                       // prologue: chunk 0
    tdm_load_2d((uint32_t)(uintptr_t)&At[0][0][0], abase, 32, 16, 32, 16, 784);
    tdm_load_2d((uint32_t)(uintptr_t)&Wt[0][0][0][0], wfrag, 2048, 1, 2048, 1, 2048);
  }
  for (int kc = 0; kc < 25; ++kc) {
    if (wv == 0) {
      if (kc + 1 < 25) {                               // prefetch next chunk
        const int kn = kc + 1;
        uint32_t valid = (uint32_t)(784 - kn * 32);
        if (valid > 32u) valid = 32u;
        tdm_load_2d((uint32_t)(uintptr_t)&At[kn & 1][0][0],
                    abase + kn * 32, valid, 16, 32, 16, 784);
        tdm_load_2d((uint32_t)(uintptr_t)&Wt[kn & 1][0][0][0],
                    wfrag + (size_t)kn * 2048, 2048, 1, 2048, 1, 2048);
        __builtin_amdgcn_s_wait_tensorcnt(2);          // chunk kc landed
      } else {
        __builtin_amdgcn_s_wait_tensorcnt(0);
      }
    }
    __syncthreads();                                   // data visible to all waves
    const int m = lane & 15, g = lane >> 4;
    const v8h* ar = (const v8h*)&At[kc & 1][m][0];
    v8h lo = ar[2 * g];                                // K = 16g..16g+7
    v8h hi = ar[2 * g + 1];                            // K = 16g+8..16g+15
    v16h a = __builtin_shufflevector(lo, hi, 0, 1, 2, 3, 4, 5, 6, 7,
                                     8, 9, 10, 11, 12, 13, 14, 15);
    v16h bm = *(const v16h*)&Wt[kc & 1][wv][lane][0];
    acc = __builtin_amdgcn_wmma_f32_16x16x32_f16(false, a, false, bm,
                                                 (short)0, acc, false, false);
    __syncthreads();                                   // reads done before overwrite
  }
#else
  // Fallback: manual pre-swizzled staging (no TDM builtin on this toolchain).
  __shared__ __align__(32) _Float16 As[32][16];
  __shared__ __align__(32) _Float16 Ws[4][32][16];
  (void)wfrag;
  for (int kc = 0; kc < 25; ++kc) {
    __syncthreads();
    for (int e = tid; e < 512; e += 128) {             // A chunk 16x32
      int m = e >> 5, kk = e & 31;
      int k = kc * 32 + kk;
      float v = (k < 784) ? (float)a3[(mtile * 16 + m) * 784 + k] : 0.0f;
      As[m + ((kk >> 4) << 4)][kk & 15] = (_Float16)v;
    }
    for (int e = tid; e < 2048; e += 128) {            // W chunk 32x64
      int kk = e >> 6, n = e & 63;
      int k = kc * 32 + kk;
      float v = (k < 784) ? w1[n * 784 + k] : 0.0f;
      Ws[n >> 4][(n & 15) + ((kk >> 4) << 4)][kk & 15] = (_Float16)v;
    }
    __syncthreads();
    v16h a  = *(const v16h*)(&As[lane][0]);
    v16h bm = *(const v16h*)(&Ws[wv][lane][0]);
    acc = __builtin_amdgcn_wmma_f32_16x16x32_f16(false, a, false, bm,
                                                 (short)0, acc, false, false);
  }
#endif
  const int n  = wv * 16 + (lane & 15);
  const float bb = b1[n];
  const int mb = (lane >> 4) * 8;
#pragma unroll
  for (int v = 0; v < 8; ++v) {
    int row = mtile * 16 + mb + v;
    float o = acc[v] + bb;
    out[row * 64 + n] = o > 0.0f ? o : 0.0f;
  }
}

// ---------------------------------------------------------------------------
// Quantum circuit helpers: 16 complex amplitudes fully registerized.
// Wire w lives on bit (3-w) of the flat state index.
// ---------------------------------------------------------------------------
template <int BIT>
__device__ __forceinline__ void apply1q(float sr[16], float si[16],
                                        float a_r, float a_i, float b_r, float b_i,
                                        float c_r, float c_i, float d_r, float d_i) {
#pragma unroll
  for (int i0 = 0; i0 < 16; ++i0) {
    if (i0 & (1 << BIT)) continue;
    const int i1 = i0 | (1 << BIT);
    float xr = sr[i0], xi = si[i0], yr = sr[i1], yi = si[i1];
    sr[i0] = a_r * xr - a_i * xi + b_r * yr - b_i * yi;
    si[i0] = a_r * xi + a_i * xr + b_r * yi + b_i * yr;
    sr[i1] = c_r * xr - c_i * xi + d_r * yr - d_i * yi;
    si[i1] = c_r * xi + c_i * xr + d_r * yi + d_i * yr;
  }
}

__device__ __forceinline__ void apply1q_w(int wire, float sr[16], float si[16],
                                          float a_r, float a_i, float b_r, float b_i,
                                          float c_r, float c_i, float d_r, float d_i) {
  switch (wire) {
    case 0: apply1q<3>(sr, si, a_r, a_i, b_r, b_i, c_r, c_i, d_r, d_i); break;
    case 1: apply1q<2>(sr, si, a_r, a_i, b_r, b_i, c_r, c_i, d_r, d_i); break;
    case 2: apply1q<1>(sr, si, a_r, a_i, b_r, b_i, c_r, c_i, d_r, d_i); break;
    default: apply1q<0>(sr, si, a_r, a_i, b_r, b_i, c_r, c_i, d_r, d_i); break;
  }
}

template <int CB, int TB>
__device__ __forceinline__ void cnot_t(float sr[16], float si[16]) {
#pragma unroll
  for (int i = 0; i < 16; ++i) {
    if (!(i & (1 << CB))) continue;
    if (i & (1 << TB)) continue;
    const int j = i | (1 << TB);
    float tr = sr[i]; sr[i] = sr[j]; sr[j] = tr;
    float ti = si[i]; si[i] = si[j]; si[j] = ti;
  }
}

__device__ __forceinline__ void cnot_w(int w0, int w1, float sr[16], float si[16]) {
  switch (w0 * 4 + w1) {
    case 1:  cnot_t<3, 2>(sr, si); break;
    case 2:  cnot_t<3, 1>(sr, si); break;
    case 3:  cnot_t<3, 0>(sr, si); break;
    case 4:  cnot_t<2, 3>(sr, si); break;
    case 6:  cnot_t<2, 1>(sr, si); break;
    case 7:  cnot_t<2, 0>(sr, si); break;
    case 8:  cnot_t<1, 3>(sr, si); break;
    case 9:  cnot_t<1, 2>(sr, si); break;
    case 11: cnot_t<1, 0>(sr, si); break;
    case 12: cnot_t<0, 3>(sr, si); break;
    case 13: cnot_t<0, 2>(sr, si); break;
    case 14: cnot_t<0, 1>(sr, si); break;
    default: break;
  }
}

// ---------------------------------------------------------------------------
// Kernel 5: FC2 + full 4-qubit circuit + PauliZ expvals + per-block
// deterministic tree reduction of (sum, sumsq) for batch-norm.
// Gate structure arrives by value (kernarg), theta from random_params.
// ---------------------------------------------------------------------------
__global__ void __launch_bounds__(256) k5_fc2_circuit(const float* __restrict__ fc1out,
                                                      const float* __restrict__ fc2_w,
                                                      const float* __restrict__ fc2_b,
                                                      const float* __restrict__ rparams,
                                                      const float* __restrict__ rxyz,
                                                      GateList g,
                                                      float* __restrict__ expvals,
                                                      float* __restrict__ partials) {
  const int tid = threadIdx.x;
  const int b   = blockIdx.x * 256 + tid;              // 8192 samples

  float ang[4];
#pragma unroll
  for (int j = 0; j < 4; ++j) {
    float s = fc2_b[j];
    for (int k = 0; k < 64; ++k) s += fc1out[b * 64 + k] * fc2_w[j * 64 + k];
    ang[j] = s;
  }

  float sr[16], si[16];
#pragma unroll
  for (int i = 0; i < 16; ++i) { sr[i] = 0.0f; si[i] = 0.0f; }
  sr[0] = 1.0f;

  // Angle encoding: RY(angle_w) per wire.
#pragma unroll
  for (int w = 0; w < 4; ++w) {
    float c = __builtin_cosf(ang[w] * 0.5f), s = __builtin_sinf(ang[w] * 0.5f);
    apply1q_w(w, sr, si, c, 0, -s, 0, s, 0, c, 0);
  }
  // Random layer (fixed structure, 50 gates).
  for (int i = 0; i < N_OPS; ++i) {
    int ty = g.t[i];
    if (ty == 3) {
      cnot_w(g.w0[i], g.w1[i], sr, si);
    } else {
      float th = rparams[i];
      float c = __builtin_cosf(th * 0.5f), s = __builtin_sinf(th * 0.5f);
      if (ty == 0)      apply1q_w(g.w0[i], sr, si, c, 0, 0, -s, 0, -s, c, 0);   // RX
      else if (ty == 1) apply1q_w(g.w0[i], sr, si, c, 0, -s, 0, s, 0, c, 0);    // RY
      else              apply1q_w(g.w0[i], sr, si, c, -s, 0, 0, 0, 0, c, s);    // RZ
    }
  }
  // Trailing trainable gates.
  {
    float c = __builtin_cosf(rxyz[0] * 0.5f), s = __builtin_sinf(rxyz[0] * 0.5f);
    apply1q<3>(sr, si, c, 0, 0, -s, 0, -s, c, 0);       // RX wire0
    c = __builtin_cosf(rxyz[1] * 0.5f); s = __builtin_sinf(rxyz[1] * 0.5f);
    apply1q<2>(sr, si, c, 0, -s, 0, s, 0, c, 0);        // RY wire1
    c = __builtin_cosf(rxyz[2] * 0.5f); s = __builtin_sinf(rxyz[2] * 0.5f);
    apply1q<1>(sr, si, c, -s, 0, 0, 0, 0, c, s);        // RZ wire2
    cnot_t<3, 0>(sr, si);                               // CNOT(0,3)
    const float r = 0.70710678118654752f;               // H wire3
    apply1q<0>(sr, si, r, 0, r, 0, r, 0, -r, 0);
  }

  // PauliZ expvals.
  float e[4];
#pragma unroll
  for (int w = 0; w < 4; ++w) {
    float s = 0.0f;
#pragma unroll
    for (int i = 0; i < 16; ++i) {
      float p = sr[i] * sr[i] + si[i] * si[i];
      s += ((i >> (3 - w)) & 1) ? -p : p;
    }
    e[w] = s;
    expvals[b * 4 + w] = s;
  }

  // Deterministic per-block reduction of sum / sumsq for each channel.
  __shared__ float rs[4][256];
  __shared__ float rq[4][256];
#pragma unroll
  for (int c = 0; c < 4; ++c) { rs[c][tid] = e[c]; rq[c][tid] = e[c] * e[c]; }
  __syncthreads();
  for (int s = 128; s > 0; s >>= 1) {
    if (tid < s) {
#pragma unroll
      for (int c = 0; c < 4; ++c) {
        rs[c][tid] += rs[c][tid + s];
        rq[c][tid] += rq[c][tid + s];
      }
    }
    __syncthreads();
  }
  if (tid < 4) {
    partials[blockIdx.x * 8 + tid]     = rs[tid][0];
    partials[blockIdx.x * 8 + 4 + tid] = rq[tid][0];
  }
}

// Kernel 6: combine 32 block partials -> stats[8] (deterministic, 1 block).
__global__ void k6_stats(const float* __restrict__ partials, float* __restrict__ stats) {
  int t = threadIdx.x;
  if (t < 8) {
    float a = 0.0f;
    for (int b = 0; b < 32; ++b) a += partials[b * 8 + t];
    stats[t] = a;
  }
}

// Kernel 7: batch-norm (training mode, biased var) -> f32 output.
__global__ void __launch_bounds__(256) k7_bn(const float* __restrict__ expvals,
                                             const float* __restrict__ stats,
                                             const float* __restrict__ gamma,
                                             const float* __restrict__ beta,
                                             float* __restrict__ out) {
  int i = blockIdx.x * 256 + threadIdx.x;
  if (i >= BATCH * 4) return;
  int c = i & 3;
  float mean = stats[c] * (1.0f / BATCH);
  float var  = stats[4 + c] * (1.0f / BATCH) - mean * mean;
  float e    = expvals[i];
  out[i] = gamma[c] * (e - mean) / __builtin_sqrtf(var + 1e-5f) + beta[c];
}

// ---------------------------------------------------------------------------
// Host: numpy-legacy MT19937 replication (RandomState(42)) for the fixed gate
// structure; deterministic, no device traffic (passed by value as kernarg).
// ---------------------------------------------------------------------------
namespace {
struct MT {
  uint32_t mt[624]; int mti;
  void seed(uint32_t s) {
    mt[0] = s;
    for (int i = 1; i < 624; ++i)
      mt[i] = 1812433253u * (mt[i - 1] ^ (mt[i - 1] >> 30)) + (uint32_t)i;
    mti = 624;
  }
  uint32_t next() {
    if (mti >= 624) {
      for (int k = 0; k < 624; ++k) {
        uint32_t y = (mt[k] & 0x80000000u) | (mt[(k + 1) % 624] & 0x7fffffffu);
        mt[k] = mt[(k + 397) % 624] ^ (y >> 1) ^ ((y & 1u) ? 0x9908b0dfu : 0u);
      }
      mti = 0;
    }
    uint32_t y = mt[mti++];
    y ^= y >> 11; y ^= (y << 7) & 0x9d2c5680u; y ^= (y << 15) & 0xefc60000u; y ^= y >> 18;
    return y;
  }
  // legacy randint(4): masked next_uint64 (hi drawn first, mask=3 -> low word)
  int randint4() { (void)next(); return (int)(next() & 3u); }
  // legacy random_interval: 32-bit masked rejection
  uint32_t interval(uint32_t max) {
    if (max == 0) return 0;
    uint32_t mask = max;
    mask |= mask >> 1; mask |= mask >> 2; mask |= mask >> 4;
    mask |= mask >> 8; mask |= mask >> 16;
    uint32_t v;
    do { v = next() & mask; } while (v > max);
    return v;
  }
  void choice2of4(int& a, int& b) {     // permutation(4)[:2] via Fisher-Yates
    int arr[4] = {0, 1, 2, 3};
    for (int i = 3; i >= 1; --i) {
      int j = (int)interval((uint32_t)i);
      int t = arr[i]; arr[i] = arr[j]; arr[j] = t;
    }
    a = arr[0]; b = arr[1];
  }
};

void make_gates(GateList& g) {
  MT rng; rng.seed(42u);
  for (int i = 0; i < N_OPS; ++i) {
    int t = rng.randint4();
    if (t == 3) {
      int a, b; rng.choice2of4(a, b);
      g.t[i] = 3; g.w0[i] = a; g.w1[i] = b;
    } else {
      int w = rng.randint4();  // randint(N_WIRES), N_WIRES==4
      g.t[i] = t; g.w0[i] = w; g.w1[i] = -1;
    }
  }
}
}  // namespace

extern "C" void kernel_launch(void* const* d_in, const int* in_sizes, int n_in,
                              void* d_out, int out_size, void* d_ws, size_t ws_size,
                              hipStream_t stream) {
  (void)in_sizes; (void)n_in; (void)out_size; (void)ws_size;
  const float* x       = (const float*)d_in[0];
  const float* conv1_w = (const float*)d_in[1];
  const float* conv1_b = (const float*)d_in[2];
  const float* conv2_w = (const float*)d_in[3];
  const float* conv2_b = (const float*)d_in[4];
  const float* fc1_w   = (const float*)d_in[5];
  const float* fc1_b   = (const float*)d_in[6];
  const float* fc2_w   = (const float*)d_in[7];
  const float* fc2_b   = (const float*)d_in[8];
  const float* rparams = (const float*)d_in[9];
  const float* rxyz    = (const float*)d_in[10];
  const float* bn_g    = (const float*)d_in[11];
  const float* bn_b    = (const float*)d_in[12];
  float* out = (float*)d_out;

  // Workspace layout (bytes, 256-aligned).
  char* ws = (char*)d_ws;
  const size_t o_h1    = 0;                          // f16 [8192][8][14][14]
  const size_t o_h2    = o_h1 + 25690112;            // f16 [8192][16][14][14]
  const size_t o_a3    = o_h2 + 51380224;            // f16 [8192][784]
  const size_t o_fc1   = o_a3 + 12845056;            // f32 [8192][64]
  const size_t o_exp   = o_fc1 + 2097152;            // f32 [8192][4]
  const size_t o_part  = o_exp + 131072;             // f32 [32][8]
  const size_t o_stats = o_part + 1024;              // f32 [8]
  const size_t o_wf    = o_stats + 256;              // f16 [25][4][32][16]
  _Float16* h1     = (_Float16*)(ws + o_h1);
  _Float16* h2     = (_Float16*)(ws + o_h2);
  _Float16* a3     = (_Float16*)(ws + o_a3);
  float*    fc1out = (float*)(ws + o_fc1);
  float*    expv   = (float*)(ws + o_exp);
  float*    part   = (float*)(ws + o_part);
  float*    stats  = (float*)(ws + o_stats);
  _Float16* wfrag  = (_Float16*)(ws + o_wf);

  GateList gates;
  make_gates(gates);

  k0_wswz<<<200, 256, 0, stream>>>(fc1_w, wfrag);
  k1_conv1<<<50176, 256, 0, stream>>>(x, conv1_w, conv1_b, h1);
  k2_conv2_wmma<<<25088, 128, 0, stream>>>(h1, conv2_w, conv2_b, h2);
  k3_pool<<<25088, 256, 0, stream>>>(h2, a3);
  k4_fc1_wmma<<<512, 128, 0, stream>>>(a3, fc1_w, wfrag, fc1_b, fc1out);
  k5_fc2_circuit<<<32, 256, 0, stream>>>(fc1out, fc2_w, fc2_b, rparams, rxyz,
                                         gates, expv, part);
  k6_stats<<<1, 32, 0, stream>>>(part, stats);
  k7_bn<<<128, 256, 0, stream>>>(expv, stats, bn_g, bn_b, out);
}
